// KerasDense_32263794328408
// MI455X (gfx1250) — compile-verified, matrix operand order checked by reading, not run on hardware
//
#include <hip/hip_runtime.h>

// ---------------------------------------------------------------------------
// TT-matrix dense layer on MI455X (gfx1250):
//   1) reconstruct M = TT(G0..G3) via staged contractions (tiny FLOPs),
//      writing M directly in WMMA-B fragment-packed bf16 layout
//   2) pack x into WMMA-A fragment-packed bf16 layout
//   3) one 4096x4096x4096 bf16 GEMM with v_wmma_f32_16x16x32_bf16,
//      async global->LDS staging (ASYNCcnt), 256x128 block, 64x64 per wave,
//      fused bias + relu epilogue
// ---------------------------------------------------------------------------

typedef __attribute__((ext_vector_type(16))) __bf16 v16bf;
typedef __attribute__((ext_vector_type(8)))  float  v8f;

#define NFULL 4096
#define KT_COUNT 128   // 4096 / 32

static __device__ __forceinline__ unsigned short f32_to_bf16(float f) {
    unsigned int u = __float_as_uint(f);
    u += 0x7FFFu + ((u >> 16) & 1u);   // round-to-nearest-even
    return (unsigned short)(u >> 16);
}

// Per-lane K index inside a 32-deep WMMA 16-bit fragment (ISA 7.12.2):
// lanes 0-15: e=0..7 -> K 0..7, e=8..15 -> K 16..23 ; lanes 16-31: +8
static __device__ __forceinline__ int kmap(int lane, int e) {
    return e + (e & 8) + ((lane & 16) >> 1);
}

// --------------------------- stage 1: T2 -----------------------------------
// T2[i1,j1,i2,j2,r2] = sum_r1 G0[0,i1,j1,r1] * G1[r1,i2,j2,r2]   (32768 elems)
__global__ void build_t2(const float* __restrict__ g0, const float* __restrict__ g1,
                         float* __restrict__ t2) {
    int idx = blockIdx.x * 256 + threadIdx.x;    // i1*4096 + j1*512 + i2*64 + j2*8 + r2
    int r2 = idx & 7;
    int j2 = (idx >> 3) & 7;
    int i2 = (idx >> 6) & 7;
    int j1 = (idx >> 9) & 7;
    int i1 = (idx >> 12) & 7;
    float s = 0.f;
#pragma unroll
    for (int r1 = 0; r1 < 8; ++r1)
        s += g0[(i1 * 8 + j1) * 8 + r1] * g1[((r1 * 8 + i2) * 8 + j2) * 8 + r2];
    t2[idx] = s;
}

// --------------------------- stage 2: T3 -----------------------------------
// T3[(i1,j1,i2,j2), i3, j3, r3] = sum_r2 T2[(i1,j1,i2,j2), r2] * G2[r2,i3,j3,r3]
__global__ void build_t3(const float* __restrict__ t2, const float* __restrict__ g2,
                         float* __restrict__ t3) {
    int idx = blockIdx.x * 256 + threadIdx.x;    // base*512 + i3*64 + j3*8 + r3
    int r3 = idx & 7;
    int j3 = (idx >> 3) & 7;
    int i3 = (idx >> 6) & 7;
    int base = idx >> 9;                          // (i1,j1,i2,j2)
    float s = 0.f;
#pragma unroll
    for (int r2 = 0; r2 < 8; ++r2)
        s += t2[base * 8 + r2] * g2[((r2 * 8 + i3) * 8 + j3) * 8 + r3];
    t3[idx] = s;
}

// --------------------------- stage 3: M -> Bpack ---------------------------
// M[k,n] = sum_r3 T3[(i1,j1,i2,j2), i3, j3, r3] * G3[r3,i4,j4,0]
// stored as bf16 in WMMA-B fragment layout:
//   Bpack[((nt*128 + kt)*32 + lane)*16 + e] = M[kt*32 + kmap(lane,e), nt*16 + lane%16]
__global__ void build_bpack(const float* __restrict__ t3, const float* __restrict__ g3,
                            unsigned short* __restrict__ bpack) {
    unsigned int q = blockIdx.x * 256 + threadIdx.x;
    int e    = q & 15;
    int lane = (q >> 4) & 31;
    int kt   = (q >> 9) & 127;
    int nt   = q >> 16;
    int k = kt * 32 + kmap(lane, e);
    int n = nt * 16 + (lane & 15);
    int i1 = (k >> 9) & 7, i2 = (k >> 6) & 7, i3 = (k >> 3) & 7, i4 = k & 7;
    int j1 = (n >> 9) & 7, j2 = (n >> 6) & 7, j3 = (n >> 3) & 7, j4 = n & 7;
    const float* t3p = t3 + ((size_t)(i1 * 512 + j1 * 64 + i2 * 8 + j2) * 512
                             + i3 * 64 + j3 * 8);
    float s = 0.f;
#pragma unroll
    for (int r3 = 0; r3 < 8; ++r3)
        s += t3p[r3] * g3[r3 * 64 + i4 * 8 + j4];
    bpack[q] = f32_to_bf16(s);
}

// --------------------------- stage 4: x -> Apack ---------------------------
// Apack[((mt*128 + kt)*32 + lane)*16 + e] = x[mt*16 + lane%16, kt*32 + kmap(lane,e)]
__global__ void pack_a(const float* __restrict__ x, unsigned short* __restrict__ apack) {
    unsigned int q = blockIdx.x * 256 + threadIdx.x;
    int e    = q & 15;
    int lane = (q >> 4) & 31;
    int kt   = (q >> 9) & 127;
    int mt   = q >> 16;
    int m = mt * 16 + (lane & 15);
    int k = kt * 32 + kmap(lane, e);
    apack[q] = f32_to_bf16(x[(size_t)m * NFULL + k]);
}

// --------------------------- stage 5: GEMM ---------------------------------
// C[4096,4096] = A(bf16) x B(bf16), bias + relu. Block 256x128, 8 wave32s,
// each wave 64x64 (4x4 WMMA tiles). Async global->LDS double buffer, 48 KB.
__global__ __launch_bounds__(256) void tt_gemm(const unsigned short* __restrict__ apack,
                                               const unsigned short* __restrict__ bpack,
                                               const float* __restrict__ bias,
                                               float* __restrict__ out) {
    // [buf][tile 0..15 = A (256 rows), 16..23 = B (128 cols)][512 bf16]
    __shared__ __align__(32) unsigned short lds[2][24][512];   // 48 KB

    const int t    = threadIdx.x;
    const int wave = t >> 5;
    const int lane = t & 31;
    const int wr   = wave & 3;   // 4 row-waves of 64 rows
    const int wc   = wave >> 2;  // 2 col-waves of 64 cols
    const int blockMt = blockIdx.y * 16;  // 16-row A tiles
    const int blockNt = blockIdx.x * 8;   // 16-col B tiles

    // cooperative async-copy roles: 6 x 16B chunks per thread per K-step
    const char* gsrc[6];
    unsigned    loff[6];
#pragma unroll
    for (int j = 0; j < 6; ++j) {
        const int ch   = j * 256 + t;        // chunk 0..1535 (16B each)
        const int tile = ch >> 6;            // 64 chunks (1 KB) per tile
        const int offB = (ch & 63) * 16;     // byte offset within tile
        const unsigned short* base;
        int srcTile;
        if (tile < 16) { base = apack; srcTile = blockMt + tile; }
        else           { base = bpack; srcTile = blockNt + tile - 16; }
        gsrc[j] = (const char*)base + (size_t)srcTile * 131072 + offB; // 128*512*2 B per tile-row
        loff[j] = (unsigned)(uintptr_t)(&lds[0][tile][0]) + offB;      // LDS byte offset
    }

// ASYNCcnt-tracked DMA: per-lane 16B global -> LDS, no VGPR data round-trip
#define STAGE(KT, BUF) do {                                                     \
        _Pragma("unroll")                                                       \
        for (int j = 0; j < 6; ++j) {                                           \
            asm volatile("global_load_async_to_lds_b128 %0, %1, off"            \
                         :: "v"(loff[j] + (unsigned)(BUF) * 24576u),            \
                            "v"(gsrc[j] + (size_t)(KT) * 1024)                  \
                         : "memory");                                           \
        }                                                                       \
    } while (0)

    v8f acc[4][4];
#pragma unroll
    for (int r = 0; r < 4; ++r)
#pragma unroll
        for (int c = 0; c < 4; ++c) acc[r][c] = v8f{};

    STAGE(0, 0);
    for (int kt = 0; kt < KT_COUNT; ++kt) {
        const int buf = kt & 1;
        asm volatile("s_wait_asynccnt 0x0" ::: "memory");  // my DMA into buf done
        __syncthreads();                                   // everyone's DMA done
        if (kt + 1 < KT_COUNT) STAGE(kt + 1, buf ^ 1);

        v16bf af[4], bg[4];
#pragma unroll
        for (int r = 0; r < 4; ++r)
            af[r] = *(const v16bf*)(&lds[buf][wr * 4 + r][lane * 16]);
#pragma unroll
        for (int c = 0; c < 4; ++c)
            bg[c] = *(const v16bf*)(&lds[buf][16 + wc * 4 + c][lane * 16]);

#pragma unroll
        for (int r = 0; r < 4; ++r)
#pragma unroll
            for (int c = 0; c < 4; ++c)
                acc[r][c] = __builtin_amdgcn_wmma_f32_16x16x32_bf16(
                    false, af[r], false, bg[c], (short)0, acc[r][c], false, false);
    }
#undef STAGE

    // epilogue: C/D layout — VGPR g -> row g (lanes<16) / g+8 (lanes>=16), col = lane%16
    const int mBase  = blockIdx.y * 256 + wr * 64;
    const int nBase  = blockIdx.x * 128 + wc * 64;
    const int col0   = lane & 15;
    const int rowOff = (lane >> 4) << 3;
#pragma unroll
    for (int r = 0; r < 4; ++r) {
#pragma unroll
        for (int c = 0; c < 4; ++c) {
            const int colg = nBase + c * 16 + col0;
            const float bv = bias[colg];
#pragma unroll
            for (int g = 0; g < 8; ++g) {
                const int rowg = mBase + r * 16 + rowOff + g;
                float v = acc[r][c][g] + bv;
                out[(size_t)rowg * NFULL + colg] = v > 0.f ? v : 0.f;
            }
        }
    }
}

// ---------------------------------------------------------------------------
extern "C" void kernel_launch(void* const* d_in, const int* in_sizes, int n_in,
                              void* d_out, int out_size, void* d_ws, size_t ws_size,
                              hipStream_t stream) {
    const float* x    = (const float*)d_in[0];
    const float* g0   = (const float*)d_in[1];
    const float* g1   = (const float*)d_in[2];
    const float* g2   = (const float*)d_in[3];
    const float* g3   = (const float*)d_in[4];
    const float* bias = (const float*)d_in[5];
    float* out = (float*)d_out;

    char* ws = (char*)d_ws;
    unsigned short* apack = (unsigned short*)(ws);                               // 32 MB
    unsigned short* bpack = (unsigned short*)(ws + (size_t)32 * 1024 * 1024);    // 32 MB
    float* t2 = (float*)(ws + (size_t)64 * 1024 * 1024);                         // 128 KB
    float* t3 = (float*)(ws + (size_t)64 * 1024 * 1024 + 256 * 1024);            // 8 MB

    build_t2<<<128, 256, 0, stream>>>(g0, g1, t2);
    build_t3<<<8192, 256, 0, stream>>>(t2, g2, t3);
    build_bpack<<<65536, 256, 0, stream>>>(t3, g3, bpack);
    pack_a<<<65536, 256, 0, stream>>>(x, apack);
    dim3 grid(32, 16);
    tt_gemm<<<grid, 256, 0, stream>>>(apack, bpack, bias, out);
}
